// IntegerDiscreteFlow_74182675136614
// MI455X (gfx1250) — matrix operand order repaired and checked by reading, not compiled, not run on hardware
//
#include <hip/hip_runtime.h>
#include <math.h>

typedef _Float16 v4h  __attribute__((ext_vector_type(4)));
typedef _Float16 v8h  __attribute__((ext_vector_type(8)));
typedef _Float16 v16h __attribute__((ext_vector_type(16)));
typedef float    v8f  __attribute__((ext_vector_type(8)));

#define PA 40   // LDS row pitch (f16) for x32-k tiles; 80B rows keep 16B alignment

// ---------------------------------------------------------------------------
// helpers
// ---------------------------------------------------------------------------
__device__ __forceinline__ unsigned lds_off(const void* p) {
    return (unsigned)(uintptr_t)p;   // low 32 bits of generic LDS address = LDS byte offset
}

// Async copy of a (NGRAN*64)-row x 32-k f16 tile into LDS (row pitch PA).
// NGRAN granules of 8 f16 per thread = NGRAN async ops per wave.
template<int NGRAN>
__device__ __forceinline__ void stage_async(const _Float16* __restrict__ G,
                                            int ldRow, int baseRow, int k0,
                                            _Float16* ldsBase, int t)
{
    #pragma unroll
    for (int s = 0; s < NGRAN; s++) {
        int q   = t + s * 256;
        int row = q >> 2;
        int kk  = (q & 3) * 8;
        unsigned loff = lds_off(ldsBase + row * PA + kk);
        unsigned goff = (unsigned)((((size_t)(baseRow + row)) * ldRow + k0 + kk) * 2);
        asm volatile("global_load_async_to_lds_b128 %0, %1, %2"
                     :: "v"(loff), "v"(goff), "s"(G) : "memory");
    }
}

// Documented 16-bit WMMA fragment layouts:
// A 16x32: lane<16 -> K 0..7 & 16..23 ; lane>=16 -> K 8..15 & 24..31
// B 32x16: lane<16 -> K 0..7 & 16..23 ; lane>=16 -> K 8..15 & 24..31 (n-major LDS row)
template<int NI, int NJ, int WNW>
__device__ __forceinline__ void wmma_tiles(const _Float16* lA, const _Float16* lB,
                                           int wm, int wn, int row16, int half,
                                           v8f acc[NI][NJ])
{
    v16h afr[NI];
    #pragma unroll
    for (int i = 0; i < NI; i++) {
        int row = wm * 64 + i * 16 + row16;
        v8h lo = *(const v8h*)(lA + row * PA + half * 8);
        v8h hi = *(const v8h*)(lA + row * PA + 16 + half * 8);
        afr[i] = __builtin_shufflevector(lo, hi, 0,1,2,3,4,5,6,7,8,9,10,11,12,13,14,15);
    }
    v16h bfr[NJ];
    #pragma unroll
    for (int j = 0; j < NJ; j++) {
        int col = wn * WNW + j * 16 + row16;
        v8h lo = *(const v8h*)(lB + col * PA + half * 8);
        v8h hi = *(const v8h*)(lB + col * PA + 16 + half * 8);
        bfr[j] = __builtin_shufflevector(lo, hi, 0,1,2,3,4,5,6,7,8,9,10,11,12,13,14,15);
    }
    #pragma unroll
    for (int i = 0; i < NI; i++)
        #pragma unroll
        for (int j = 0; j < NJ; j++)
            acc[i][j] = __builtin_amdgcn_wmma_f32_16x16x32_f16(
                false, afr[i], false, bfr[j], (short)0, acc[i][j], false, false);
}

// Epilogue: MODE 0 -> leaky_relu, f16 store; MODE 1 -> reversed coupling update
// of f32 state (+ optional f16 mirror).
template<int MODE, int NJ, int WNW>
__device__ __forceinline__ void epilogue(v8f acc[4][NJ], void* out, _Float16* mirror,
                                         const float* bias, int m0, int n0, int N,
                                         int wm, int wn, int row16, int half)
{
    #pragma unroll
    for (int i = 0; i < 4; i++) {
        #pragma unroll
        for (int j = 0; j < NJ; j++) {
            int gmBase = m0 + wm * 64 + i * 16 + half * 8;
            int gn     = n0 + wn * WNW + j * 16 + row16;
            float bv = bias[gn];
            #pragma unroll
            for (int r = 0; r < 8; r++) {
                int gm = gmBase + r;
                float v = acc[i][j][r] + bv;
                if (MODE == 0) {
                    v = v > 0.0f ? v : 0.01f * v;
                    ((_Float16*)out)[(size_t)gm * N + gn] = (_Float16)v;
                } else {
                    float* bp = (float*)out;
                    size_t idx = (size_t)gm * 512 + (size_t)(511 - gn);
                    float nv = bp[idx] + rintf(v);
                    bp[idx] = nv;
                    if (mirror) mirror[idx] = (_Float16)nv;
                }
            }
        }
    }
}

// ---------------------------------------------------------------------------
// FAST path: block tile 256x128, 8 waves as 4x2 of 64x64 wave tiles.
// f16 A + pre-converted f16 W^T, async global->LDS, double buffered.
// ---------------------------------------------------------------------------
template<int MODE>
__global__ __launch_bounds__(256)
void idf_gemm_fast(const _Float16* __restrict__ A,   // M x K f16 row-major
                   const _Float16* __restrict__ Wt,  // N x K f16 row-major
                   const float* __restrict__ bias,
                   void* __restrict__ out,
                   _Float16* __restrict__ outMirror,
                   int K, int N)
{
    __shared__ alignas(16) _Float16 ldsA[2][256 * PA];   // 2 x 20 KB
    __shared__ alignas(16) _Float16 ldsB[2][128 * PA];   // 2 x 10 KB

    const int t    = threadIdx.x;
    const int lane = t & 31;
    const int wid  = t >> 5;
    const int wm   = wid >> 1;          // 0..3 : 64-row slab
    const int wn   = wid & 1;           // 0..1 : 64-col slab
    const int m0   = blockIdx.y * 256;
    const int n0   = blockIdx.x * 128;
    const int row16 = lane & 15;
    const int half  = lane >> 4;

    v8f acc[4][4];
    {
        v8f z = {};
        for (int i = 0; i < 4; i++)
            for (int j = 0; j < 4; j++) acc[i][j] = z;
    }

    const int niter = K >> 5;
    // prologue: stage chunk 0 into buffer 0 (4 + 2 = 6 async ops per wave)
    stage_async<4>(A,  K, m0, 0, &ldsA[0][0], t);
    stage_async<2>(Wt, K, n0, 0, &ldsB[0][0], t);

    for (int i = 0; i < niter; i++) {
        if (i + 1 < niter) {
            int k0n = (i + 1) << 5;
            int nb  = (i + 1) & 1;
            stage_async<4>(A,  K, m0, k0n, &ldsA[nb][0], t);
            stage_async<2>(Wt, K, n0, k0n, &ldsB[nb][0], t);
            // 12 outstanding; wait until only the 6 just-issued remain
            asm volatile("s_wait_asynccnt 0x6" ::: "memory");
        } else {
            asm volatile("s_wait_asynccnt 0x0" ::: "memory");
        }
        __syncthreads();                        // tile i visible to all waves
        wmma_tiles<4, 4, 64>(&ldsA[i & 1][0], &ldsB[i & 1][0],
                             wm, wn, row16, half, acc);
        __syncthreads();                        // everyone done reading buffer i
    }

    epilogue<MODE, 4, 64>(acc, out, outMirror, bias, m0, n0, N, wm, wn, row16, half);
}

// ---------------------------------------------------------------------------
// BASIC fallback path: block 128x128, waves 2x4 of
// 64x32 tiles; f32/f16 A, f32 W with in-kernel convert+transpose.
// ---------------------------------------------------------------------------
template<int A_F16, int MODE>
__global__ __launch_bounds__(256)
void idf_gemm_basic(const void* __restrict__ Aptr,
                    const float* __restrict__ W,
                    const float* __restrict__ bias,
                    void* __restrict__ out,
                    int K, int N)
{
    __shared__ alignas(16) _Float16 ldsA[128 * PA];
    __shared__ alignas(16) _Float16 ldsB[128 * PA];

    const int t    = threadIdx.x;
    const int lane = t & 31;
    const int wid  = t >> 5;
    const int wm   = wid >> 2;
    const int wn   = wid & 3;
    const int m0   = blockIdx.y * 128;
    const int n0   = blockIdx.x * 128;
    const int row16 = lane & 15;
    const int half  = lane >> 4;

    v8f acc[4][2];
    {
        v8f z = {};
        for (int i = 0; i < 4; i++)
            for (int j = 0; j < 2; j++) acc[i][j] = z;
    }

    for (int k0 = 0; k0 < K; k0 += 32) {
        if (A_F16) {
            const _Float16* A = (const _Float16*)Aptr;
            #pragma unroll
            for (int s = 0; s < 2; s++) {
                int q   = t + s * 256;
                int row = q >> 2;
                int kk  = (q & 3) * 8;
                v8h v = *(const v8h*)(A + (size_t)(m0 + row) * K + k0 + kk);
                *(v8h*)&ldsA[row * PA + kk] = v;
            }
        } else {
            const float* A = (const float*)Aptr;
            #pragma unroll
            for (int s = 0; s < 4; s++) {
                int q   = t + s * 256;
                int row = q >> 3;
                int kk  = (q & 7) * 4;
                float4 v = *(const float4*)(A + (size_t)(m0 + row) * K + k0 + kk);
                v4h h;
                h[0] = (_Float16)v.x; h[1] = (_Float16)v.y;
                h[2] = (_Float16)v.z; h[3] = (_Float16)v.w;
                *(v4h*)&ldsA[row * PA + kk] = h;
            }
        }
        #pragma unroll
        for (int s = 0; s < 4; s++) {
            int q  = t + s * 256;
            int kk = q >> 5;
            int nn = (q & 31) * 4;
            float4 v = *(const float4*)(W + (size_t)(k0 + kk) * N + n0 + nn);
            ldsB[(nn + 0) * PA + kk] = (_Float16)v.x;
            ldsB[(nn + 1) * PA + kk] = (_Float16)v.y;
            ldsB[(nn + 2) * PA + kk] = (_Float16)v.z;
            ldsB[(nn + 3) * PA + kk] = (_Float16)v.w;
        }
        __syncthreads();
        wmma_tiles<4, 2, 32>(&ldsA[0], &ldsB[0], wm, wn, row16, half, acc);
        __syncthreads();
    }

    epilogue<MODE, 2, 32>(acc, out, (_Float16*)nullptr, bias, m0, n0, N,
                          wm, wn, row16, half);
}

// ---------------------------------------------------------------------------
// Weight pre-conversion: W (K x N f32) -> Wt (N x K f16), per step (blockIdx.z)
// ---------------------------------------------------------------------------
__global__ __launch_bounds__(256)
void idf_wt_convert(const float* __restrict__ W, _Float16* __restrict__ Wt,
                    int K, int N)
{
    __shared__ _Float16 tile[64][72];   // [n][k], padded
    W  += (size_t)blockIdx.z * K * N;
    Wt += (size_t)blockIdx.z * K * N;
    const int k0 = blockIdx.y * 64;
    const int n0 = blockIdx.x * 64;
    const int t  = threadIdx.x;

    #pragma unroll
    for (int s = 0; s < 4; s++) {
        int q  = t + s * 256;
        int kk = q >> 4;
        int nn = (q & 15) * 4;
        float4 v = *(const float4*)(W + (size_t)(k0 + kk) * N + n0 + nn);
        tile[nn + 0][kk] = (_Float16)v.x;
        tile[nn + 1][kk] = (_Float16)v.y;
        tile[nn + 2][kk] = (_Float16)v.z;
        tile[nn + 3][kk] = (_Float16)v.w;
    }
    __syncthreads();
    #pragma unroll
    for (int s = 0; s < 2; s++) {
        int q  = t + s * 256;
        int nn = q >> 3;
        int kk = (q & 7) * 8;
        v8h v = *(const v8h*)&tile[nn][kk];
        *(v8h*)(Wt + (size_t)(n0 + nn) * K + k0 + kk) = v;
    }
}

// ---------------------------------------------------------------------------
// State init + final density reduction
// ---------------------------------------------------------------------------
__global__ void idf_init(const float* __restrict__ x,
                         float* __restrict__ bufY, float* __restrict__ bufX)
{
    size_t i = (size_t)blockIdx.x * 256 + threadIdx.x;
    size_t b = i >> 9;
    int    j = (int)(i & 511);
    bufY[i] = x[b * 1024 + j];
    bufX[i] = x[b * 1024 + (1023 - j)];
}

__global__ void idf_init_m(const float* __restrict__ x,
                           float* __restrict__ bufY, float* __restrict__ bufX,
                           _Float16* __restrict__ bufYh, _Float16* __restrict__ bufXh)
{
    size_t i = (size_t)blockIdx.x * 256 + threadIdx.x;
    size_t b = i >> 9;
    int    j = (int)(i & 511);
    float a = x[b * 1024 + j];
    float c = x[b * 1024 + (1023 - j)];
    bufY[i] = a;  bufYh[i] = (_Float16)a;
    bufX[i] = c;  bufXh[i] = (_Float16)c;
}

__device__ __forceinline__ float log_sig(float u) {
    return fminf(u, 0.0f) - log1pf(expf(-fabsf(u)));
}

__global__ void idf_reduce(const float* __restrict__ bufY, const float* __restrict__ bufX,
                           const float* __restrict__ mean, const float* __restrict__ logscale,
                           float* __restrict__ partial)
{
    __shared__ float sm[256];
    const size_t total = (size_t)8192 * 1024;
    float sum = 0.0f;
    for (size_t i = (size_t)blockIdx.x * 256 + threadIdx.x; i < total;
         i += (size_t)gridDim.x * 256) {
        size_t b = i >> 10;
        int    j = (int)(i & 1023);
        float z = (j < 512) ? bufY[b * 512 + j] : bufX[b * 512 + (1023 - j)];
        float sc = expf(logscale[j]);
        float mu = mean[j];
        float la = log_sig((z + 0.5f - mu) / sc);
        float lb = log_sig((z - 0.5f - mu) / sc);
        sum += la + logf(1.0f - expf(lb - la) + 1e-8f);
    }
    sm[threadIdx.x] = sum;
    __syncthreads();
    for (int s = 128; s > 0; s >>= 1) {
        if ((int)threadIdx.x < s) sm[threadIdx.x] += sm[threadIdx.x + s];
        __syncthreads();
    }
    if (threadIdx.x == 0) partial[blockIdx.x] = sm[0];
}

__global__ void idf_final(const float* __restrict__ partial, int n, float* __restrict__ out)
{
    __shared__ float sm[256];
    float s = 0.0f;
    for (int i = threadIdx.x; i < n; i += 256) s += partial[i];
    sm[threadIdx.x] = s;
    __syncthreads();
    for (int k = 128; k > 0; k >>= 1) {
        if ((int)threadIdx.x < k) sm[threadIdx.x] += sm[threadIdx.x + k];
        __syncthreads();
    }
    if (threadIdx.x == 0) out[0] = -sm[0] / 8192.0f;
}

// ---------------------------------------------------------------------------
extern "C" void kernel_launch(void* const* d_in, const int* in_sizes, int n_in,
                              void* d_out, int out_size, void* d_ws, size_t ws_size,
                              hipStream_t stream)
{
    const float* x   = (const float*)d_in[0];
    const float* W1  = (const float*)d_in[1];
    const float* b1  = (const float*)d_in[2];
    const float* W2  = (const float*)d_in[3];
    const float* b2  = (const float*)d_in[4];
    const float* W3  = (const float*)d_in[5];
    const float* b3  = (const float*)d_in[6];
    const float* mean     = (const float*)d_in[7];
    const float* logscale = (const float*)d_in[8];

    const int B = 8192, N = 2048, F = 8, d2 = 512;
    const size_t MB = (size_t)1 << 20;
    char* ws = (char*)d_ws;

    float* bufX = (float*)ws;                    // 16 MB : xa_{-1},xa_1,...,xa_7
    float* bufY = (float*)(ws + 16 * MB);        // 16 MB : xa_0,xa_2,...,xa_8

    const size_t FULL_NEED = 208 * MB + 8192;

    if (ws_size >= FULL_NEED) {
        _Float16* bufXh = (_Float16*)(ws + 32 * MB);
        _Float16* bufYh = (_Float16*)(ws + 40 * MB);
        _Float16* h1    = (_Float16*)(ws + 48 * MB);
        _Float16* h2    = (_Float16*)(ws + 80 * MB);
        _Float16* Wt1   = (_Float16*)(ws + 112 * MB);  // 8 x (2048 x 512)
        _Float16* Wt2   = (_Float16*)(ws + 128 * MB);  // 8 x (2048 x 2048)
        _Float16* Wt3   = (_Float16*)(ws + 192 * MB);  // 8 x (512 x 2048)
        float*    partial = (float*)(ws + 208 * MB);

        idf_wt_convert<<<dim3(N / 64, d2 / 64, F), 256, 0, stream>>>(W1, Wt1, d2, N);
        idf_wt_convert<<<dim3(N / 64, N / 64, F), 256, 0, stream>>>(W2, Wt2, N, N);
        idf_wt_convert<<<dim3(d2 / 64, N / 64, F), 256, 0, stream>>>(W3, Wt3, N, d2);

        idf_init_m<<<(B * d2) / 256, 256, 0, stream>>>(x, bufY, bufX, bufYh, bufXh);

        for (int s = 0; s < F; s++) {
            const _Float16* Ah   = (s & 1) ? bufXh : bufYh;
            float*          Pbuf = (s & 1) ? bufY : bufX;
            _Float16*       Pmh  = (s & 1) ? bufYh : bufXh;

            dim3 g1(N / 128, B / 256);     // (16, 32)
            idf_gemm_fast<0><<<g1, 256, 0, stream>>>(
                Ah, Wt1 + (size_t)s * N * d2, b1 + (size_t)s * N, h1, nullptr, d2, N);
            idf_gemm_fast<0><<<g1, 256, 0, stream>>>(
                h1, Wt2 + (size_t)s * N * N, b2 + (size_t)s * N, h2, nullptr, N, N);
            dim3 g3(d2 / 128, B / 256);    // (4, 32)
            idf_gemm_fast<1><<<g3, 256, 0, stream>>>(
                h2, Wt3 + (size_t)s * d2 * N, b3 + (size_t)s * d2, Pbuf, Pmh, N, d2);
        }

        idf_reduce<<<1024, 256, 0, stream>>>(bufY, bufX, mean, logscale, partial);
        idf_final<<<1, 256, 0, stream>>>(partial, 1024, (float*)d_out);
    } else {
        _Float16* h1 = (_Float16*)(ws + 32 * MB);
        _Float16* h2 = (_Float16*)(ws + 64 * MB);
        float* partial = (float*)(ws + 96 * MB);

        idf_init<<<(B * d2) / 256, 256, 0, stream>>>(x, bufY, bufX);

        for (int s = 0; s < F; s++) {
            const void* Acur = (s & 1) ? (const void*)bufX : (const void*)bufY;
            float*      Pbuf = (s & 1) ? bufY : bufX;

            dim3 g1(N / 128, B / 128);
            idf_gemm_basic<0, 0><<<g1, 256, 0, stream>>>(
                Acur, W1 + (size_t)s * d2 * N, b1 + (size_t)s * N, h1, d2, N);
            idf_gemm_basic<1, 0><<<g1, 256, 0, stream>>>(
                h1, W2 + (size_t)s * N * N, b2 + (size_t)s * N, h2, N, N);
            dim3 g3(d2 / 128, B / 128);
            idf_gemm_basic<1, 1><<<g3, 256, 0, stream>>>(
                h2, W3 + (size_t)s * N * d2, b3 + (size_t)s * d2, Pbuf, N, d2);
        }

        idf_reduce<<<1024, 256, 0, stream>>>(bufY, bufX, mean, logscale, partial);
        idf_final<<<1, 256, 0, stream>>>(partial, 1024, (float*)d_out);
    }
}
